// DynamicConv_62689342653031
// MI455X (gfx1250) — matrix operand order repaired and checked by reading, not compiled, not run on hardware
//
#include <hip/hip_runtime.h>
#include <hip/hip_bf16.h>
#include <math.h>

typedef __attribute__((ext_vector_type(16))) _Float16 v16h;
typedef __attribute__((ext_vector_type(8)))  float    v8f;
typedef __attribute__((ext_vector_type(2)))  float    fv2;
typedef __attribute__((__vector_size__(4 * sizeof(int)))) int v4i_t;

#define L_LEN   262144
#define B_SZ    32
#define MAXLEN  262142            // L - 3 + 1
#define OUT_CH  4
#define TCHUNK  4096
#define CHUNKS_PER_B (L_LEN / TCHUNK)   // 64
#define NBLK    (B_SZ * CHUNKS_PER_B)   // 2048

// workspace layout (float offsets)
#define WS_POOLED 0       // 256 floats
#define WS_KVAR   256     // 5 variants * 4 ch * 12 taps = 240 floats
#define WS_SCALE  512     // 4
#define WS_SHIFT  516     // 4
#define WS_PART   1024    // NBLK * 8 partial sums

// address-space-typed pointers for the gfx1250 async-to-LDS builtin
#define GV4I(p) ((__attribute__((address_space(1))) v4i_t*)(p))
#define LV4I(p) ((__attribute__((address_space(3))) v4i_t*)(p))

__device__ __forceinline__ void wait_async_zero() {
#if __has_builtin(__builtin_amdgcn_s_wait_asynccnt)
    __builtin_amdgcn_s_wait_asynccnt(0);
#else
    asm volatile("s_wait_asynccnt 0" ::: "memory");
#endif
}

// ---------------------------------------------------------------------------
// Kernel 1: adaptive-avg-pool of batch 0: pooled[j] = mean(x[0, j*1024 .. +1023])
// ---------------------------------------------------------------------------
__global__ void pool_kernel(const float* __restrict__ x, float* __restrict__ ws) {
    __shared__ float red[256];
    const int j   = blockIdx.x;      // 0..255
    const int tid = threadIdx.x;     // 0..255
    const float4* xv = (const float4*)(x + (size_t)j * 1024);
    float4 v = xv[tid];
    red[tid] = v.x + v.y + v.z + v.w;
    __syncthreads();
    for (int s = 128; s > 0; s >>= 1) {
        if (tid < s) red[tid] += red[tid + s];
        __syncthreads();
    }
    if (tid == 0) ws[WS_POOLED + j] = red[0] * (1.0f / 1024.0f);
}

// ---------------------------------------------------------------------------
// Kernel 2: MLP (WMMA f16) + softmax + build effective kernel table.
// One wave (32 threads). h = relu(W1 p + b1); y = softmax(W2 h + b2)[0..11];
// Kvar[v][c][j] = sum_i { KS[i] <= thr[v] && j < KS[i] } s_i * k_i[c][j]
// ---------------------------------------------------------------------------
__global__ void mlp_kernel(float* __restrict__ ws,
                           const float* __restrict__ w1, const float* __restrict__ b1,
                           const float* __restrict__ w2, const float* __restrict__ b2,
                           const float* k0, const float* k1, const float* k2,
                           const float* k3, const float* k4, const float* k5,
                           const float* k6, const float* k7, const float* k8,
                           const float* k9, const float* k10, const float* k11) {
    __shared__ _Float16 pbuf[256];
    __shared__ _Float16 hbuf[128];
    __shared__ float    ybuf[16];

    const int lane = threadIdx.x;        // 0..31
    const int m_l  = lane & 15;
    const int half = lane >> 4;

    for (int i = lane; i < 256; i += 32)
        pbuf[i] = (_Float16)ws[WS_POOLED + i];
    __syncthreads();

    // ---- layer 1: h[128] = relu(W1[128,256] * p + b1) ----
    for (int mt = 0; mt < 8; ++mt) {
        v8f acc = {};
        const int row = mt * 16 + m_l;
        for (int kc = 0; kc < 8; ++kc) {
            v16h a, b;
#pragma unroll
            for (int v = 0; v < 8; ++v) {
                int kb = kc * 32 + ((v < 4) ? v * 2 : 16 + (v - 4) * 2) + half * 8;
                a[2 * v]     = (_Float16)w1[row * 256 + kb];
                a[2 * v + 1] = (_Float16)w1[row * 256 + kb + 1];
            }
#pragma unroll
            for (int i = 0; i < 16; ++i)       // B: p replicated across all 16 cols
                b[i] = pbuf[kc * 32 + half * 8 + i + ((i >= 8) ? 8 : 0)];
            acc = __builtin_amdgcn_wmma_f32_16x16x32_f16(
                false, a, false, b, (short)0, acc, false, false);
        }
        if (lane == 0)
            for (int v = 0; v < 8; ++v)
                hbuf[mt * 16 + v] = (_Float16)fmaxf(acc[v] + b1[mt * 16 + v], 0.0f);
        if (lane == 16)
            for (int v = 0; v < 8; ++v)
                hbuf[mt * 16 + 8 + v] = (_Float16)fmaxf(acc[v] + b1[mt * 16 + 8 + v], 0.0f);
        __syncthreads();
    }

    // ---- layer 2: y[12] = W2[12,128] * h + b2 ----
    {
        v8f acc = {};
        for (int kc = 0; kc < 4; ++kc) {
            v16h a, b;
#pragma unroll
            for (int v = 0; v < 8; ++v) {
                int kb = kc * 32 + ((v < 4) ? v * 2 : 16 + (v - 4) * 2) + half * 8;
                _Float16 e0 = (m_l < 12) ? (_Float16)w2[m_l * 128 + kb]     : (_Float16)0.0f;
                _Float16 e1 = (m_l < 12) ? (_Float16)w2[m_l * 128 + kb + 1] : (_Float16)0.0f;
                a[2 * v]     = e0;
                a[2 * v + 1] = e1;
            }
#pragma unroll
            for (int i = 0; i < 16; ++i)
                b[i] = hbuf[kc * 32 + half * 8 + i + ((i >= 8) ? 8 : 0)];
            acc = __builtin_amdgcn_wmma_f32_16x16x32_f16(
                false, a, false, b, (short)0, acc, false, false);
        }
        if (lane == 0)
            for (int v = 0; v < 8; ++v) ybuf[v] = acc[v] + b2[v];
        if (lane == 16)
            for (int v = 0; v < 8; ++v)
                if (8 + v < 12) ybuf[8 + v] = acc[v] + b2[8 + v];
        __syncthreads();
    }

    // ---- softmax + effective kernel table (scalar on lane 0, tiny) ----
    if (lane == 0) {
        float s[12];
        float mx = ybuf[0];
        for (int i = 1; i < 12; ++i) mx = fmaxf(mx, ybuf[i]);
        float den = 0.0f;
        for (int i = 0; i < 12; ++i) { s[i] = __expf(ybuf[i] - mx); den += s[i]; }
        float inv = 1.0f / den;
        for (int i = 0; i < 12; ++i) s[i] *= inv;

        const float* kp[12] = {k0,k1,k2,k3,k4,k5,k6,k7,k8,k9,k10,k11};
        const int ks[12]  = {3,3,3,5,5,7,7,7,9,9,11,11};
        const int thr[5]  = {3,5,7,9,11};
        for (int v = 0; v < 5; ++v)
            for (int c = 0; c < OUT_CH; ++c)
                for (int j = 0; j < 12; ++j) {
                    float acc = 0.0f;
                    if (j < 11) {
#pragma unroll
                        for (int i = 0; i < 12; ++i)
                            if (ks[i] <= thr[v] && j < ks[i])
                                acc += s[i] * kp[i][c * ks[i] + j];
                    }
                    ws[WS_KVAR + (v * 4 + c) * 12 + j] = acc;
                }
    }
}

// ---------------------------------------------------------------------------
// Shared tile loader: stage TCHUNK+16 floats of one row into LDS.
// Fast path uses gfx1250 async global->LDS b128 DMA (ASYNCcnt tracked).
// ---------------------------------------------------------------------------
__device__ __forceinline__ void load_tile(const float* __restrict__ xrow, int t0,
                                          float* tile, int tid) {
    if (t0 + TCHUNK + 16 <= L_LEN) {
#if __has_builtin(__builtin_amdgcn_global_load_async_to_lds_b128)
        {
            __attribute__((address_space(1))) v4i_t* src = GV4I(xrow + t0 + tid * 4);
            __attribute__((address_space(3))) v4i_t* dst = LV4I(tile + tid * 4);
            __builtin_amdgcn_global_load_async_to_lds_b128(src, dst, 0,     0);
            __builtin_amdgcn_global_load_async_to_lds_b128(src, dst, 16384, 0);
            __builtin_amdgcn_global_load_async_to_lds_b128(src, dst, 32768, 0);
            __builtin_amdgcn_global_load_async_to_lds_b128(src, dst, 49152, 0);
            if (tid < 4)
                __builtin_amdgcn_global_load_async_to_lds_b128(
                    GV4I(xrow + t0 + 4096 + tid * 4), LV4I(tile + 4096 + tid * 4), 0, 0);
        }
        if (t0 + 2 * TCHUNK <= L_LEN)
            __builtin_prefetch(xrow + t0 + TCHUNK + tid * 16, 0, 0); // global_prefetch_b8
        wait_async_zero();
#else
        const float4* src = (const float4*)(xrow + t0);
        float4*       dst = (float4*)tile;
#pragma unroll
        for (int i = 0; i < 4; ++i) dst[tid + i * 256] = src[tid + i * 256];
        if (tid < 4) dst[1024 + tid] = src[1024 + tid];
        if (t0 + 2 * TCHUNK <= L_LEN)
            __builtin_prefetch(xrow + t0 + TCHUNK + tid * 16, 0, 0);
#endif
    } else {
        for (int i = tid; i < TCHUNK + 16; i += 256) {
            int idx = t0 + i;
            tile[i] = (idx < L_LEN) ? xrow[idx] : 0.0f;
        }
    }
}

__device__ __forceinline__ int variant_of(int t) {
    int d = L_LEN - t;   // >= 3 for all valid t
    return (d >= 11) ? 4 : (d >= 9) ? 3 : (d >= 7) ? 2 : (d >= 5) ? 1 : 0;
}

// ---------------------------------------------------------------------------
// Kernel 3: pass 1 — streaming conv, per-channel sum / sumsq block partials
// ---------------------------------------------------------------------------
__global__ void conv_stats_kernel(const float* __restrict__ x, float* __restrict__ ws) {
    __shared__ __align__(16) float tile[TCHUNK + 16];
    __shared__ float kv[240];
    __shared__ float red[256];

    const int tid   = threadIdx.x;
    const int b     = blockIdx.x / CHUNKS_PER_B;
    const int chunk = blockIdx.x % CHUNKS_PER_B;
    const int t0    = chunk * TCHUNK;
    const float* xrow = x + (size_t)b * L_LEN;

    for (int i = tid; i < 240; i += 256) kv[i] = ws[WS_KVAR + i];
    load_tile(xrow, t0, tile, tid);
    __syncthreads();

    float Kf[OUT_CH][11];
#pragma unroll
    for (int c = 0; c < OUT_CH; ++c)
#pragma unroll
        for (int j = 0; j < 11; ++j) Kf[c][j] = kv[(16 + c) * 12 + j];

    float sum[OUT_CH] = {0, 0, 0, 0};
    float sq[OUT_CH]  = {0, 0, 0, 0};

    for (int g = 0; g < 4; ++g) {
        const int lt = g * 1024 + tid * 4;
        const int t  = t0 + lt;
        if (t <= L_LEN - 14) {                       // all 4 positions full-kernel
            float xs[14];
#pragma unroll
            for (int j = 0; j < 14; ++j) xs[j] = tile[lt + j];
#pragma unroll
            for (int c = 0; c < OUT_CH; ++c) {
                float a0 = 0, a1 = 0, a2 = 0, a3 = 0;
#pragma unroll
                for (int j = 0; j < 11; ++j) {
                    float w = Kf[c][j];
                    a0 = fmaf(w, xs[j],     a0);
                    a1 = fmaf(w, xs[j + 1], a1);
                    a2 = fmaf(w, xs[j + 2], a2);
                    a3 = fmaf(w, xs[j + 3], a3);
                }
                sum[c] += (a0 + a1) + (a2 + a3);
                sq[c]  += (a0 * a0 + a1 * a1) + (a2 * a2 + a3 * a3);
            }
        } else {                                     // boundary tail (rare)
            for (int u = 0; u < 4; ++u) {
                int tt = t + u;
                if (tt >= MAXLEN) continue;
                int v = variant_of(tt);
                for (int c = 0; c < OUT_CH; ++c) {
                    float a = 0.0f;
#pragma unroll
                    for (int j = 0; j < 11; ++j)
                        a = fmaf(kv[(v * 4 + c) * 12 + j], tile[lt + u + j], a);
                    sum[c] += a;
                    sq[c]  += a * a;
                }
            }
        }
    }

    for (int comp = 0; comp < 8; ++comp) {
        red[tid] = (comp < 4) ? sum[comp] : sq[comp - 4];
        __syncthreads();
        for (int s = 128; s > 0; s >>= 1) {
            if (tid < s) red[tid] += red[tid + s];
            __syncthreads();
        }
        if (tid == 0) ws[WS_PART + (size_t)blockIdx.x * 8 + comp] = red[0];
        __syncthreads();
    }
}

// ---------------------------------------------------------------------------
// Kernel 4: deterministic final reduce -> per-channel BN scale/shift
// ---------------------------------------------------------------------------
__global__ void finalize_kernel(float* __restrict__ ws,
                                const float* __restrict__ gamma,
                                const float* __restrict__ beta) {
    __shared__ float red[256];
    __shared__ float tot[8];
    const int tid = threadIdx.x;
    for (int comp = 0; comp < 8; ++comp) {
        float l = 0.0f;
        for (int i = tid; i < NBLK; i += 256) l += ws[WS_PART + (size_t)i * 8 + comp];
        red[tid] = l;
        __syncthreads();
        for (int s = 128; s > 0; s >>= 1) {
            if (tid < s) red[tid] += red[tid + s];
            __syncthreads();
        }
        if (tid == 0) tot[comp] = red[0];
        __syncthreads();
    }
    if (tid < OUT_CH) {
        const float N = (float)B_SZ * (float)MAXLEN;   // exact in fp32 (< 2^24)
        float mean = tot[tid] / N;
        float var  = tot[tid + 4] / N - mean * mean;
        float inv  = rsqrtf(var + 1e-5f);
        float sc   = gamma[tid] * inv;
        ws[WS_SCALE + tid] = sc;
        ws[WS_SHIFT + tid] = beta[tid] - mean * sc;
    }
}

// ---------------------------------------------------------------------------
// Kernel 5: pass 2 — recompute conv, apply BN + ReLU, write output.
// Output is write-once: non-temporal stores keep x resident in L2.
// ---------------------------------------------------------------------------
__global__ void conv_write_kernel(const float* __restrict__ x,
                                  const float* __restrict__ ws,
                                  float* __restrict__ out) {
    __shared__ __align__(16) float tile[TCHUNK + 16];
    __shared__ float kv[240];

    const int tid   = threadIdx.x;
    const int b     = blockIdx.x / CHUNKS_PER_B;
    const int chunk = blockIdx.x % CHUNKS_PER_B;
    const int t0    = chunk * TCHUNK;
    const float* xrow = x + (size_t)b * L_LEN;

    for (int i = tid; i < 240; i += 256) kv[i] = ws[WS_KVAR + i];
    load_tile(xrow, t0, tile, tid);
    __syncthreads();

    float Kf[OUT_CH][11];
    float sc[OUT_CH], sh[OUT_CH];
#pragma unroll
    for (int c = 0; c < OUT_CH; ++c) {
        sc[c] = ws[WS_SCALE + c];
        sh[c] = ws[WS_SHIFT + c];
#pragma unroll
        for (int j = 0; j < 11; ++j) Kf[c][j] = kv[(16 + c) * 12 + j];
    }

    for (int g = 0; g < 4; ++g) {
        const int lt = g * 1024 + tid * 4;
        const int t  = t0 + lt;
        if (t <= L_LEN - 14) {
            float xs[14];
#pragma unroll
            for (int j = 0; j < 14; ++j) xs[j] = tile[lt + j];
#pragma unroll
            for (int c = 0; c < OUT_CH; ++c) {
                float a0 = 0, a1 = 0, a2 = 0, a3 = 0;
#pragma unroll
                for (int j = 0; j < 11; ++j) {
                    float w = Kf[c][j];
                    a0 = fmaf(w, xs[j],     a0);
                    a1 = fmaf(w, xs[j + 1], a1);
                    a2 = fmaf(w, xs[j + 2], a2);
                    a3 = fmaf(w, xs[j + 3], a3);
                }
                a0 = fmaxf(fmaf(a0, sc[c], sh[c]), 0.0f);
                a1 = fmaxf(fmaf(a1, sc[c], sh[c]), 0.0f);
                a2 = fmaxf(fmaf(a2, sc[c], sh[c]), 0.0f);
                a3 = fmaxf(fmaf(a3, sc[c], sh[c]), 0.0f);
                float* op = out + ((size_t)(b * OUT_CH + c)) * MAXLEN + t;  // 8B aligned
                fv2 r01 = {a0, a1};
                fv2 r23 = {a2, a3};
                __builtin_nontemporal_store(r01, (fv2*)op);
                __builtin_nontemporal_store(r23, (fv2*)op + 1);
            }
        } else {
            for (int u = 0; u < 4; ++u) {
                int tt = t + u;
                if (tt >= MAXLEN) continue;
                int v = variant_of(tt);
                for (int c = 0; c < OUT_CH; ++c) {
                    float a = 0.0f;
#pragma unroll
                    for (int j = 0; j < 11; ++j)
                        a = fmaf(kv[(v * 4 + c) * 12 + j], tile[lt + u + j], a);
                    out[((size_t)(b * OUT_CH + c)) * MAXLEN + tt] =
                        fmaxf(fmaf(a, sc[c], sh[c]), 0.0f);
                }
            }
        }
    }
}

// ---------------------------------------------------------------------------
extern "C" void kernel_launch(void* const* d_in, const int* in_sizes, int n_in,
                              void* d_out, int out_size, void* d_ws, size_t ws_size,
                              hipStream_t stream) {
    const float* x     = (const float*)d_in[0];
    const float* w1    = (const float*)d_in[1];
    const float* b1    = (const float*)d_in[2];
    const float* w2    = (const float*)d_in[3];
    const float* b2    = (const float*)d_in[4];
    const float* gamma = (const float*)d_in[5];
    const float* beta  = (const float*)d_in[6];
    float* ws  = (float*)d_ws;
    float* out = (float*)d_out;

    pool_kernel<<<256, 256, 0, stream>>>(x, ws);
    mlp_kernel<<<1, 32, 0, stream>>>(ws, w1, b1, w2, b2,
        (const float*)d_in[7],  (const float*)d_in[8],  (const float*)d_in[9],
        (const float*)d_in[10], (const float*)d_in[11], (const float*)d_in[12],
        (const float*)d_in[13], (const float*)d_in[14], (const float*)d_in[15],
        (const float*)d_in[16], (const float*)d_in[17], (const float*)d_in[18]);
    conv_stats_kernel<<<NBLK, 256, 0, stream>>>(x, ws);
    finalize_kernel<<<1, 256, 0, stream>>>(ws, gamma, beta);
    conv_write_kernel<<<NBLK, 256, 0, stream>>>(x, ws, out);
}
